// TransformerEncoderLayer_66151086293335
// MI455X (gfx1250) — matrix operand order repaired and checked by reading, not compiled, not run on hardware
//
#include <hip/hip_runtime.h>
#include <hip/hip_bf16.h>

// ---------------- problem constants ----------------
#define SEQ    1024
#define BATCH  8
#define DMODEL 512
#define NHEAD  8
#define DK     64
#define DFF    2048
#define MROWS  (SEQ * BATCH)   // 8192 rows in [B,S] order
#define SPITCH 1028            // LDS score pitch (conflict-free for phase-3 reads)

typedef _Float16 h16 __attribute__((ext_vector_type(16)));
typedef _Float16 h8  __attribute__((ext_vector_type(8)));
typedef float    f8  __attribute__((ext_vector_type(8)));

// ---------------- WMMA helpers ----------------
// A fragment (16x32 f16, row-major source, arbitrary lda):
// lane holds row m0 + lane%16; element i holds k = k0 + (lane/16)*16 + i (contiguous)
__device__ __forceinline__ h16 load_a(const _Float16* __restrict__ A, int lda,
                                      int m0, int k0, int lane) {
  const int m = m0 + (lane & 15);
  const int half = lane >> 4;
  const _Float16* p = A + (size_t)m * lda + k0 + half * 16;
  return *(const h16*)(p);
}

// B fragment (32x16 f16) sourced from row-major W[n][k] (i.e. computing A @ W^T):
// lane holds col n0 + lane%16; element i holds k = k0 + (lane/16)*16 + i (contiguous)
__device__ __forceinline__ h16 load_b(const _Float16* __restrict__ W, int ldb,
                                      int n0, int k0, int lane) {
  const int n = n0 + (lane & 15);
  const int half = lane >> 4;
  return *(const h16*)(W + (size_t)n * ldb + k0 + half * 16);
}

__device__ __forceinline__ f8 wmma32(h16 a, h16 b, f8 c) {
  return __builtin_amdgcn_wmma_f32_16x16x32_f16(false, a, false, b, (short)0, c,
                                                false, false);
}

// ---------------- converts ----------------
__global__ __launch_bounds__(256) void cvt16(const float* __restrict__ in,
                                             _Float16* __restrict__ out, int n) {
  int i = blockIdx.x * 256 + threadIdx.x;
  if (i < n) out[i] = (_Float16)in[i];
}

// x16[b*S+s][d] = (f16) src[s*B+b][d]
__global__ __launch_bounds__(256) void prep_x(const float* __restrict__ src,
                                              _Float16* __restrict__ x16) {
  int i = blockIdx.x * 256 + threadIdx.x;            // over MROWS*DMODEL
  int row = i >> 9, d = i & (DMODEL - 1);
  int b = row >> 10, s = row & (SEQ - 1);
  x16[i] = (_Float16)src[((size_t)s * BATCH + b) * DMODEL + d];
}

// ---------------- generic WMMA GEMM: C = A[M,K] @ W[N,K]^T ----------------
// one wave computes a 16x64 tile; block = 8 waves stacked in M.
// MODE 0: o16 = (f16)(acc*scale)                         (Q/K projection)
// MODE 1: o16 = V transposed per head -> vt[b,h,dk,S]    (V projection)
// MODE 2: o32 = acc + src[s*B+b] (residual remap)        (fc)
// MODE 3: o16 = relu(acc + bias[n])                      (ffn1)
// MODE 4: o32 = acc + bias[n] + extra[m*D+n]             (ffn2)
template <int MODE>
__global__ __launch_bounds__(256) void gemm_f16(
    const _Float16* __restrict__ A, int lda, const _Float16* __restrict__ W,
    int ldb, int K, int ldc, _Float16* __restrict__ o16,
    float* __restrict__ o32, const float* __restrict__ bias,
    const float* __restrict__ extra, float scale) {
  const int lane = threadIdx.x & 31;
  const int wave = threadIdx.x >> 5;
  const int m0 = (blockIdx.x * 8 + wave) * 16;
  const int n0 = blockIdx.y * 64;
  f8 acc[4] = {};
  for (int k0 = 0; k0 < K; k0 += 32) {
    h16 a = load_a(A, lda, m0, k0, lane);
#pragma unroll
    for (int nt = 0; nt < 4; ++nt) {
      h16 bf = load_b(W, ldb, n0 + nt * 16, k0, lane);
      acc[nt] = wmma32(a, bf, acc[nt]);
    }
  }
  const int half = lane >> 4;
  const int nl = lane & 15;
#pragma unroll
  for (int nt = 0; nt < 4; ++nt) {
    const int n = n0 + nt * 16 + nl;
#pragma unroll
    for (int r = 0; r < 8; ++r) {
      const int m = m0 + half * 8 + r;
      float v = acc[nt][r];
      if constexpr (MODE == 0) {
        o16[(size_t)m * ldc + n] = (_Float16)(v * scale);
      } else if constexpr (MODE == 1) {
        int b = m >> 10, s = m & (SEQ - 1);
        int hh = n >> 6, j = n & (DK - 1);
        o16[((size_t)(b * NHEAD + hh) * DK + j) * SEQ + s] = (_Float16)v;
      } else if constexpr (MODE == 2) {
        int b = m >> 10, s = m & (SEQ - 1);
        o32[(size_t)m * DMODEL + n] =
            v + extra[((size_t)s * BATCH + b) * DMODEL + n];
      } else if constexpr (MODE == 3) {
        float t = v + bias[n];
        o16[(size_t)m * ldc + n] = (_Float16)(t > 0.f ? t : 0.f);
      } else {
        o32[(size_t)m * DMODEL + n] = v + bias[n] + extra[(size_t)m * DMODEL + n];
      }
    }
  }
}

// ---------------- fused attention: scores -> softmax -> P@V ----------------
// block = (b, h, 16-query tile); 256 threads (8 waves); scores strip in LDS.
__global__ __launch_bounds__(256) void attn_fused(
    const _Float16* __restrict__ q16, const _Float16* __restrict__ k16,
    const _Float16* __restrict__ vt16, const float* __restrict__ sph,
    float* __restrict__ p_out, _Float16* __restrict__ o16) {
  __shared__ float sc[16 * SPITCH];
  const int qt = blockIdx.x & 63;
  const int h = (blockIdx.x >> 6) & (NHEAD - 1);
  const int b = blockIdx.x >> 9;
  const int q0 = qt * 16;
  const int lane = threadIdx.x & 31;
  const int wave = threadIdx.x >> 5;
  const int half = lane >> 4;
  const int nl = lane & 15;

  const _Float16* Qb = q16 + (size_t)b * SEQ * DMODEL + h * DK;
  const _Float16* Kb = k16 + (size_t)b * SEQ * DMODEL + h * DK;

  // phase 1: scores tile 16 x 1024 (q pre-scaled by 1/TEMP), * sph, -> LDS
  h16 a0 = load_a(Qb, DMODEL, q0, 0, lane);
  h16 a1 = load_a(Qb, DMODEL, q0, 32, lane);
  for (int t = 0; t < 8; ++t) {
    const int n0 = wave * 128 + t * 16;
    f8 acc = {};
    acc = wmma32(a0, load_b(Kb, DMODEL, n0, 0, lane), acc);
    acc = wmma32(a1, load_b(Kb, DMODEL, n0, 32, lane), acc);
    const int n = n0 + nl;
#pragma unroll
    for (int r = 0; r < 8; ++r) {
      int ml = half * 8 + r;
      sc[ml * SPITCH + n] =
          acc[r] * sph[((size_t)b * SEQ + q0 + ml) * SEQ + n];
    }
  }
  __syncthreads();

  // phase 2: softmax, one row per 16 lanes (all 16 in the same wave)
  {
    const int row = threadIdx.x >> 4;
    const int c0 = threadIdx.x & 15;
    float* rp = sc + row * SPITCH;
    float mx = -__builtin_inff();
#pragma unroll 4
    for (int j = 0; j < 64; ++j) mx = fmaxf(mx, rp[j * 16 + c0]);
#pragma unroll
    for (int off = 8; off; off >>= 1) mx = fmaxf(mx, __shfl_xor(mx, off, 16));
    float sum = 0.f;
#pragma unroll 4
    for (int j = 0; j < 64; ++j) {
      float e = __expf(rp[j * 16 + c0] - mx);
      rp[j * 16 + c0] = e;
      sum += e;
    }
#pragma unroll
    for (int off = 8; off; off >>= 1) sum += __shfl_xor(sum, off, 16);
    float rinv = 1.f / sum;
    float* po = p_out + ((size_t)(b * NHEAD + h) * SEQ + q0 + row) * SEQ;
#pragma unroll 4
    for (int j = 0; j < 64; ++j) {
      float p = rp[j * 16 + c0] * rinv;
      rp[j * 16 + c0] = p;
      po[j * 16 + c0] = p;   // the single mandatory p_attn HBM write
    }
  }
  __syncthreads();

  // phase 3: out(16x64) = P(16x1024, LDS) @ V(1024x64); waves 0..3, one 16x16 tile each
  if (wave < 4) {
    const _Float16* Vb = vt16 + (size_t)(b * NHEAD + h) * DK * SEQ;
    f8 acc = {};
    const int m = nl;
    for (int k0 = 0; k0 < SEQ; k0 += 32) {
      h16 a;
#pragma unroll
      for (int i = 0; i < 16; ++i) {
        int k = k0 + half * 16 + i;
        a[i] = (_Float16)sc[m * SPITCH + k];
      }
      acc = wmma32(a, load_b(Vb, SEQ, wave * 16, k0, lane), acc);
    }
#pragma unroll
    for (int r = 0; r < 8; ++r) {
      int ml = half * 8 + r;
      o16[((size_t)b * SEQ + q0 + ml) * DMODEL + h * DK + wave * 16 + nl] =
          (_Float16)acc[r];
    }
  }
}

// ---------------- layer norms ----------------
__device__ __forceinline__ float block_sum(float v, float* sdata) {
  int tid = threadIdx.x;
  sdata[tid] = v;
  __syncthreads();
#pragma unroll
  for (int s = 128; s > 0; s >>= 1) {
    if (tid < s) sdata[tid] += sdata[tid + s];
    __syncthreads();
  }
  float r = sdata[0];
  __syncthreads();
  return r;
}

// t1 -> LN(ln_attn,1e-6) ; + src (remap) ; -> LN(n1,1e-5) -> h32,h16
__global__ __launch_bounds__(256) void ln_mid(
    const float* __restrict__ t1, const float* __restrict__ src,
    const float* __restrict__ g1, const float* __restrict__ b1,
    const float* __restrict__ g2, const float* __restrict__ b2,
    float* __restrict__ h32, _Float16* __restrict__ h16o) {
  __shared__ float sdata[256];
  const int row = blockIdx.x;
  const int tid = threadIdx.x;
  const int b = row >> 10, s = row & (SEQ - 1);
  const int d0 = tid * 2;
  float2 v = ((const float2*)(t1 + (size_t)row * DMODEL))[tid];
  float mu = block_sum(v.x + v.y, sdata) * (1.f / DMODEL);
  float dx = v.x - mu, dy = v.y - mu;
  float var = block_sum(dx * dx + dy * dy, sdata) * (1.f / DMODEL);
  float rstd = rsqrtf(var + 1e-6f);
  float2 sv = ((const float2*)(src + ((size_t)s * BATCH + b) * DMODEL))[tid];
  float ux = dx * rstd * g1[d0] + b1[d0] + sv.x;
  float uy = dy * rstd * g1[d0 + 1] + b1[d0 + 1] + sv.y;
  float mu2 = block_sum(ux + uy, sdata) * (1.f / DMODEL);
  float ex = ux - mu2, ey = uy - mu2;
  float var2 = block_sum(ex * ex + ey * ey, sdata) * (1.f / DMODEL);
  float rstd2 = rsqrtf(var2 + 1e-5f);
  float hx = ex * rstd2 * g2[d0] + b2[d0];
  float hy = ey * rstd2 * g2[d0 + 1] + b2[d0 + 1];
  ((float2*)(h32 + (size_t)row * DMODEL))[tid] = make_float2(hx, hy);
  h16o[(size_t)row * DMODEL + d0] = (_Float16)hx;
  h16o[(size_t)row * DMODEL + d0 + 1] = (_Float16)hy;
}

// t2 -> LN(n2,1e-5) -> y[s*B+b]
__global__ __launch_bounds__(256) void ln_out(const float* __restrict__ t2,
                                              const float* __restrict__ g,
                                              const float* __restrict__ bt,
                                              float* __restrict__ y) {
  __shared__ float sdata[256];
  const int row = blockIdx.x;
  const int tid = threadIdx.x;
  const int b = row >> 10, s = row & (SEQ - 1);
  const int d0 = tid * 2;
  float2 v = ((const float2*)(t2 + (size_t)row * DMODEL))[tid];
  float mu = block_sum(v.x + v.y, sdata) * (1.f / DMODEL);
  float dx = v.x - mu, dy = v.y - mu;
  float var = block_sum(dx * dx + dy * dy, sdata) * (1.f / DMODEL);
  float rstd = rsqrtf(var + 1e-5f);
  float ox = dx * rstd * g[d0] + bt[d0];
  float oy = dy * rstd * g[d0 + 1] + bt[d0 + 1];
  ((float2*)(y + ((size_t)s * BATCH + b) * DMODEL))[tid] = make_float2(ox, oy);
}

// ---------------- launcher ----------------
extern "C" void kernel_launch(void* const* d_in, const int* in_sizes, int n_in,
                              void* d_out, int out_size, void* d_ws,
                              size_t ws_size, hipStream_t stream) {
  (void)in_sizes; (void)n_in; (void)out_size;
  const float* src  = (const float*)d_in[0];
  const float* sph  = (const float*)d_in[1];
  const float* w_qs = (const float*)d_in[2];
  const float* w_ks = (const float*)d_in[3];
  const float* w_vs = (const float*)d_in[4];
  const float* w_fc = (const float*)d_in[5];
  const float* lag  = (const float*)d_in[6];
  const float* lab  = (const float*)d_in[7];
  const float* w1   = (const float*)d_in[8];
  const float* b1   = (const float*)d_in[9];
  const float* w2   = (const float*)d_in[10];
  const float* b2   = (const float*)d_in[11];
  const float* n1g  = (const float*)d_in[12];
  const float* n1b  = (const float*)d_in[13];
  const float* n2g  = (const float*)d_in[14];
  const float* n2b  = (const float*)d_in[15];

  char* ws = (char*)d_ws;
  size_t off = 0;
  auto alloc = [&](size_t bytes) -> void* {
    void* p = ws + off;
    off += (bytes + 255) & ~(size_t)255;
    return p;
  };
  _Float16* wq16  = (_Float16*)alloc((size_t)DMODEL * DMODEL * 2);
  _Float16* wk16  = (_Float16*)alloc((size_t)DMODEL * DMODEL * 2);
  _Float16* wv16  = (_Float16*)alloc((size_t)DMODEL * DMODEL * 2);
  _Float16* wfc16 = (_Float16*)alloc((size_t)DMODEL * DMODEL * 2);
  _Float16* w1_16 = (_Float16*)alloc((size_t)DFF * DMODEL * 2);
  _Float16* w2_16 = (_Float16*)alloc((size_t)DMODEL * DFF * 2);
  _Float16* x16   = (_Float16*)alloc((size_t)MROWS * DMODEL * 2);
  _Float16* q16   = (_Float16*)alloc((size_t)MROWS * DMODEL * 2);
  _Float16* k16   = (_Float16*)alloc((size_t)MROWS * DMODEL * 2);
  _Float16* vt16  = (_Float16*)alloc((size_t)MROWS * DMODEL * 2);
  _Float16* at16  = (_Float16*)alloc((size_t)MROWS * DMODEL * 2);
  float*    t1    = (float*)alloc((size_t)MROWS * DMODEL * 4);
  float*    h32   = (float*)alloc((size_t)MROWS * DMODEL * 4);
  _Float16* h16b  = (_Float16*)alloc((size_t)MROWS * DMODEL * 2);
  // aliases: ffh16 reuses the (dead by then) q16..at16 region (exactly 32MB);
  // t2 reuses t1 (dead after ln_mid).
  _Float16* ffh16 = q16;
  float*    t2    = t1;
  if (ws_size < off) return;  // scratch too small; nothing safe to do

  float* y_out = (float*)d_out;
  float* p_out = y_out + (size_t)MROWS * DMODEL;  // p_attn after y

  const int WDD = DMODEL * DMODEL, WFD = DFF * DMODEL;
  cvt16<<<(WDD + 255) / 256, 256, 0, stream>>>(w_qs, wq16, WDD);
  cvt16<<<(WDD + 255) / 256, 256, 0, stream>>>(w_ks, wk16, WDD);
  cvt16<<<(WDD + 255) / 256, 256, 0, stream>>>(w_vs, wv16, WDD);
  cvt16<<<(WDD + 255) / 256, 256, 0, stream>>>(w_fc, wfc16, WDD);
  cvt16<<<(WFD + 255) / 256, 256, 0, stream>>>(w1, w1_16, WFD);
  cvt16<<<(WFD + 255) / 256, 256, 0, stream>>>(w2, w2_16, WFD);
  prep_x<<<(MROWS * DMODEL) / 256, 256, 0, stream>>>(src, x16);

  dim3 gD(MROWS / 128, DMODEL / 64);   // 64 x 8
  dim3 gF(MROWS / 128, DFF / 64);      // 64 x 32
  // Q (pre-scaled by 1/sqrt(dk)=0.125), K, V(transposed store)
  gemm_f16<0><<<gD, 256, 0, stream>>>(x16, DMODEL, wq16, DMODEL, DMODEL, DMODEL,
                                      q16, nullptr, nullptr, nullptr, 0.125f);
  gemm_f16<0><<<gD, 256, 0, stream>>>(x16, DMODEL, wk16, DMODEL, DMODEL, DMODEL,
                                      k16, nullptr, nullptr, nullptr, 1.0f);
  gemm_f16<1><<<gD, 256, 0, stream>>>(x16, DMODEL, wv16, DMODEL, DMODEL, DMODEL,
                                      vt16, nullptr, nullptr, nullptr, 1.0f);
  // fused scores*sph -> softmax -> p_attn out -> P@V
  attn_fused<<<BATCH * NHEAD * (SEQ / 16), 256, 0, stream>>>(q16, k16, vt16,
                                                             sph, p_out, at16);
  // fc + residual(x)
  gemm_f16<2><<<gD, 256, 0, stream>>>(at16, DMODEL, wfc16, DMODEL, DMODEL,
                                      DMODEL, nullptr, t1, nullptr, src, 1.0f);
  // LN(ln_attn) + src + LN(n1) -> h
  ln_mid<<<MROWS, 256, 0, stream>>>(t1, src, lag, lab, n1g, n1b, h32, h16b);
  // ffn1: relu(h @ w1^T + b1)
  gemm_f16<3><<<gF, 256, 0, stream>>>(h16b, DMODEL, w1_16, DMODEL, DMODEL, DFF,
                                      ffh16, nullptr, b1, nullptr, 1.0f);
  // ffn2: + b2 + h
  gemm_f16<4><<<gD, 256, 0, stream>>>(ffh16, DFF, w2_16, DFF, DFF, DMODEL,
                                      nullptr, t2, b2, h32, 1.0f);
  // final LN(n2) -> y (remapped to [S,B,D])
  ln_out<<<MROWS, 256, 0, stream>>>(t2, n2g, n2b, y_out);
}
